// EGNN_dynamics_16862041604105
// MI455X (gfx1250) — matrix operand order repaired and checked by reading, not compiled, not run on hardware
//
#include <hip/hip_runtime.h>
#include <stdint.h>

// =====================================================================
// EGNN dynamics on MI455X (gfx1250, wave32, WMMA + async global->LDS).
//
// GEMMs: V_WMMA_F32_16X16X32_F16 (f16 A/B, f32 accum).
// Data movement: weights and h-activation tiles are staged into LDS with
// GLOBAL_LOAD_ASYNC_TO_LDS_B128 (ASYNCcnt; builtin probe-confirmed on
// this toolchain, param type v4i addrspace(1)*), falling back to uint4
// copies otherwise. h is mirrored in f16 (hf16) by its producers so
// edge/node A-tiles need no per-tile conversion.
// Segment sums use global_atomic_add_f32 (reference semantics).
// =====================================================================

#define HID     64
#define NLAYERS 4
#define NN      40000
#define NGG     4000
#define EE      640000
#define NATOMS  10
#define EPSLN   1e-5f

#define K1      132     // 2*HID + 1 + 3 (edge MLP input)
#define K1P     160     // padded to 5 K-tiles of 32
#define KH      128     // node MLP input (h | m_h)
#define WPB     2       // waves per block in WMMA kernels (64 threads)

typedef __attribute__((ext_vector_type(16))) _Float16 v16h;
typedef __attribute__((ext_vector_type(8)))  float    v8f;
typedef int v4i __attribute__((vector_size(16)));

union Frag16 { v16h h; uint4 q[2]; };

#if defined(__HIP_DEVICE_COMPILE__) && __has_builtin(__builtin_amdgcn_global_load_async_to_lds_b128)
#define HAVE_ASYNC 1
#else
#define HAVE_ASYNC 0
#endif

typedef __attribute__((address_space(1))) v4i* gasv4;   // global 16B vector ptr
typedef __attribute__((address_space(3))) v4i* lasv4;   // LDS    16B vector ptr

// 16-byte global -> LDS copy (async path when available)
__device__ __forceinline__ void async_cp16(const void* g, void* l)
{
#if HAVE_ASYNC
  __builtin_amdgcn_global_load_async_to_lds_b128(
      (gasv4)(uintptr_t)g, (lasv4)(uint32_t)(uintptr_t)l, 0, 0);
#else
  *(uint4*)l = *(const uint4*)g;
#endif
}

__device__ __forceinline__ void wait_async0()
{
#if HAVE_ASYNC
#if __has_builtin(__builtin_amdgcn_s_wait_asynccnt)
  __builtin_amdgcn_s_wait_asynccnt(0);
  asm volatile("" ::: "memory");
#else
  asm volatile("s_wait_asynccnt 0x0" ::: "memory");
#endif
#endif
}

__device__ __forceinline__ void wait_ds0()
{
#if __has_builtin(__builtin_amdgcn_s_wait_dscnt)
  __builtin_amdgcn_s_wait_dscnt(0);
  asm volatile("" ::: "memory");
#else
  asm volatile("s_wait_dscnt 0x0" ::: "memory");
#endif
}

__device__ __forceinline__ float silu_f(float v) { return v / (1.0f + __expf(-v)); }

// D(16x64,f32 in LDS) = A(16x[KT*32]) * B, B stored transposed [64][ldb].
template<int KT>
__device__ __forceinline__ void gemm16x64(const _Float16* A, int lda,
                                          const _Float16* Bt, int ldb,
                                          float* S, int lane)
{
  const int row  = lane & 15;
  const int half = lane >> 4;
  v8f acc[4];
#pragma unroll
  for (int n = 0; n < 4; ++n) {
#pragma unroll
    for (int i = 0; i < 8; ++i) acc[n][i] = 0.0f;
  }
#pragma unroll
  for (int kt = 0; kt < KT; ++kt) {
    Frag16 a;
    const int klo = kt * 32 + half * 8;                 // K {0..7|8..15}
    a.q[0] = *reinterpret_cast<const uint4*>(A + row * lda + klo);
    a.q[1] = *reinterpret_cast<const uint4*>(A + row * lda + klo + 16); // +16 halfs
#pragma unroll
    for (int n = 0; n < 4; ++n) {
      Frag16 b;
      const _Float16* bp = Bt + (n * 16 + row) * ldb + kt * 32 + half * 8;
      b.q[0] = *reinterpret_cast<const uint4*>(bp);
      b.q[1] = *reinterpret_cast<const uint4*>(bp + 16); // +16 halfs
      acc[n] = __builtin_amdgcn_wmma_f32_16x16x32_f16(
          false, a.h, false, b.h, (short)0, acc[n], false, false);
    }
  }
  // D layout: VGPR r, lanes 0-15 -> M=r, lanes 16-31 -> M=r+8; N = lane&15
#pragma unroll
  for (int n = 0; n < 4; ++n)
#pragma unroll
    for (int r = 0; r < 8; ++r)
      S[(half * 8 + r) * HID + n * 16 + row] = acc[n][r];
}

// -------- weight prep: f32 [K][64] -> f16 transposed [64][KP], zero-pad K
__global__ void k_w16(const float* __restrict__ src, _Float16* __restrict__ dst,
                      int K, int KP, int Nc)
{
  int i = blockIdx.x * blockDim.x + threadIdx.x;
  if (i >= Nc * KP) return;
  int col = i / KP, k = i - col * KP;
  dst[i] = (k < K) ? (_Float16)src[k * Nc + col] : (_Float16)0.0f;
}

// -------- h0 = [node_attrs | t[batch]] @ emb_W + emb_b ; x = positions
__global__ void k_init_h(const float* __restrict__ attrs, const float* __restrict__ t,
                         const int* __restrict__ batch, const float* __restrict__ embW,
                         const float* __restrict__ embB, const float* __restrict__ pos,
                         float* __restrict__ h, _Float16* __restrict__ hf16,
                         float* __restrict__ x)
{
  int i = blockIdx.x * blockDim.x + threadIdx.x;
  if (i >= NN * HID) return;
  int n = i / HID, o = i - n * HID;
  float acc = embB[o];
#pragma unroll
  for (int k = 0; k < NATOMS; ++k) acc += attrs[n * NATOMS + k] * embW[k * HID + o];
  acc += t[batch[n]] * embW[NATOMS * HID + o];
  h[i] = acc;
  hf16[i] = (_Float16)acc;
  if (o < 3) x[n * 3 + o] = pos[n * 3 + o];
}

// -------- per-edge statics: src-degree c, d2_0 from original positions
__global__ void k_edge_static(const int* __restrict__ ei, const float* __restrict__ pos,
                              float* __restrict__ cdeg, float* __restrict__ d20)
{
  int e = blockIdx.x * blockDim.x + threadIdx.x;
  if (e >= EE) return;
  int s = ei[e], d = ei[EE + e];
  float dx = pos[s * 3 + 0] - pos[d * 3 + 0];
  float dy = pos[s * 3 + 1] - pos[d * 3 + 1];
  float dz = pos[s * 3 + 2] - pos[d * 3 + 2];
  d20[e] = dx * dx + dy * dy + dz * dz;
  atomicAdd(&cdeg[s], 1.0f);
}

// -------- edge kernel: phi_e + phi_x + scatter-add (the WMMA hot loop)
__global__ __launch_bounds__(WPB * 32)
void k_edge(const int* __restrict__ ei, const int* __restrict__ ea,
            const float* __restrict__ d20,
            const _Float16* __restrict__ hf16, const float* __restrict__ x,
            const _Float16* __restrict__ W1t, const _Float16* __restrict__ W2t,
            const _Float16* __restrict__ XW1t,
            const float* __restrict__ eb1, const float* __restrict__ eg1, const float* __restrict__ eB1,
            const float* __restrict__ eb2, const float* __restrict__ eg2, const float* __restrict__ eB2,
            const float* __restrict__ xb1, const float* __restrict__ xg,  const float* __restrict__ xB,
            const float* __restrict__ xW2, const float* __restrict__ xb2,
            float* __restrict__ aggmh, float* __restrict__ aggmx)
{
  __shared__ __align__(16) _Float16 sW1[HID * K1P];      // 20480 B
  __shared__ __align__(16) _Float16 sW2[HID * HID];      //  8192 B
  __shared__ __align__(16) _Float16 sX1[HID * HID];      //  8192 B
  __shared__ float sPar[10 * HID + 1];                   //  2564 B
  __shared__ __align__(16) _Float16 sA[WPB * 16 * K1P];  // 10240 B
  __shared__ __align__(16) float    sS[WPB * 16 * HID];  //  8192 B

  const int tid = threadIdx.x;
  // weights: async global->LDS, 16B chunks
  for (int o = tid; o < (HID * K1P) / 8; o += WPB * 32) async_cp16(W1t + o * 8, sW1 + o * 8);
  for (int o = tid; o < (HID * HID) / 8; o += WPB * 32) {
    async_cp16(W2t + o * 8, sW2 + o * 8);
    async_cp16(XW1t + o * 8, sX1 + o * 8);
  }
  for (int i = tid; i < HID; i += WPB * 32) {
    sPar[0*HID+i]=eb1[i]; sPar[1*HID+i]=eg1[i]; sPar[2*HID+i]=eB1[i];
    sPar[3*HID+i]=eb2[i]; sPar[4*HID+i]=eg2[i]; sPar[5*HID+i]=eB2[i];
    sPar[6*HID+i]=xb1[i]; sPar[7*HID+i]=xg[i];  sPar[8*HID+i]=xB[i];
    sPar[9*HID+i]=xW2[i];
  }
  if (tid == 0) sPar[10*HID] = xb2[0];
  wait_async0();
  __syncthreads();

  const int wave = tid >> 5, lane = tid & 31;
  const int row = lane & 15, half = lane >> 4;
  const int cb = half * 32;                  // column split for LN phases
  _Float16* A = sA + wave * 16 * K1P;
  float*    S = sS + wave * 16 * HID;

  const int NT = (EE / 16) / WPB;
  for (int t0 = blockIdx.x; t0 < NT; t0 += gridDim.x) {
    const int eBase = (t0 * WPB + wave) * 16;
    const int e     = eBase + row;
    const int srcI  = ei[e];
    const int dstI  = ei[EE + e];
    __builtin_prefetch(&ei[e + (int)gridDim.x * WPB * 16], 0, 1); // global_prefetch_b8

    // ---- A tile h-part: async copy 16 rows x (h[dst]|h[src]) from hf16
    wait_ds0();   // order ASYNCcnt LDS writes vs previous iter's DScnt reads
#pragma unroll
    for (int it = 0; it < 8; ++it) {
      const int chunk = it * 32 + lane;      // 0..255 (16 rows x 16 chunks)
      const int r  = chunk >> 4;
      const int ci = chunk & 15;             // 0-7: h[dst] cols, 8-15: h[src] cols
      const int eR = eBase + r;
      const int nI = (ci < 8) ? ei[EE + eR] : ei[eR];
      async_cp16(hf16 + (size_t)nI * HID + (ci & 7) * 8,
                 A + r * K1P + ci * 8);
    }
    // ---- specials: r2 | onehot(bc) | d2_0 | zero pad
    if (half == 0) {
      const float dx = x[dstI*3+0]-x[srcI*3+0];
      const float dy = x[dstI*3+1]-x[srcI*3+1];
      const float dz = x[dstI*3+2]-x[srcI*3+2];
      const int bc = ea[e*2 + 1];
      A[row*K1P + 128] = (_Float16)(dx*dx + dy*dy + dz*dz);
      A[row*K1P + 129] = (_Float16)(bc == 0 ? 1.0f : 0.0f);
      A[row*K1P + 130] = (_Float16)(bc == 1 ? 1.0f : 0.0f);
      A[row*K1P + 131] = (_Float16)d20[e];
      for (int c = 132; c < 144; ++c) A[row*K1P + c] = (_Float16)0.0f;
    } else {
      for (int c = 144; c < K1P; ++c) A[row*K1P + c] = (_Float16)0.0f;
    }
    wait_async0();

    // ---- phi_e layer1: GEMM + bias + LN + SiLU  (EXEC full at WMMA)
    gemm16x64<5>(A, K1P, sW1, K1P, S, lane);
    {
      const float *bb=&sPar[0], *gg=&sPar[64], *BB=&sPar[128];
      float m=0.f, s2=0.f;
      for (int c=0;c<32;++c){ float y=S[row*HID+cb+c]+bb[cb+c]; m+=y; s2+=y*y; }
      m  += __shfl_xor(m, 16);
      s2 += __shfl_xor(s2, 16);
      m *= (1.0f/HID); s2 = s2*(1.0f/HID) - m*m;
      const float ri = rsqrtf(s2 + EPSLN);
      for (int c=0;c<32;++c){
        float y=(S[row*HID+cb+c]+bb[cb+c]-m)*ri*gg[cb+c]+BB[cb+c];
        A[row*K1P+cb+c] = (_Float16)silu_f(y);
      }
    }

    // ---- phi_e layer2 -> mh ; scatter-add agg_mh[dst]
    gemm16x64<2>(A, K1P, sW2, HID, S, lane);
    {
      const float *bb=&sPar[192], *gg=&sPar[256], *BB=&sPar[320];
      float m=0.f, s2=0.f;
      for (int c=0;c<32;++c){ float y=S[row*HID+cb+c]+bb[cb+c]; m+=y; s2+=y*y; }
      m  += __shfl_xor(m, 16);
      s2 += __shfl_xor(s2, 16);
      m *= (1.0f/HID); s2 = s2*(1.0f/HID) - m*m;
      const float ri = rsqrtf(s2 + EPSLN);
      for (int c=0;c<32;++c){
        float y=(S[row*HID+cb+c]+bb[cb+c]-m)*ri*gg[cb+c]+BB[cb+c];
        float v = silu_f(y);
        A[row*K1P+cb+c] = (_Float16)v;
        atomicAdd(&aggmh[(size_t)dstI*HID + cb + c], v);
      }
    }

    // ---- phi_x hidden + LN + SiLU, dot with x_W2 -> per-edge scalar
    gemm16x64<2>(A, K1P, sX1, HID, S, lane);
    {
      const float *bb=&sPar[384], *gg=&sPar[448], *BB=&sPar[512], *w2=&sPar[576];
      float m=0.f, s2=0.f;
      for (int c=0;c<32;++c){ float y=S[row*HID+cb+c]+bb[cb+c]; m+=y; s2+=y*y; }
      m  += __shfl_xor(m, 16);
      s2 += __shfl_xor(s2, 16);
      m *= (1.0f/HID); s2 = s2*(1.0f/HID) - m*m;
      const float ri = rsqrtf(s2 + EPSLN);
      float dot = 0.f;
      for (int c=0;c<32;++c){
        float y=(S[row*HID+cb+c]+bb[cb+c]-m)*ri*gg[cb+c]+BB[cb+c];
        dot += silu_f(y) * w2[cb+c];
      }
      dot += __shfl_xor(dot, 16);
      dot += sPar[640];                      // x_b2
      if (half == 0) {
        atomicAdd(&aggmx[dstI*3+0], (x[dstI*3+0]-x[srcI*3+0]) * dot);
        atomicAdd(&aggmx[dstI*3+1], (x[dstI*3+1]-x[srcI*3+1]) * dot);
        atomicAdd(&aggmx[dstI*3+2], (x[dstI*3+2]-x[srcI*3+2]) * dot);
      }
    }
  }
}

// -------- node kernel: h += phi_h([h|m_h]);  x += m_x / c
__global__ __launch_bounds__(WPB * 32)
void k_node(const float* __restrict__ aggmh, const float* __restrict__ aggmx,
            const float* __restrict__ cdeg,
            const _Float16* __restrict__ HW1t, const _Float16* __restrict__ HW2t,
            const float* __restrict__ hb1, const float* __restrict__ hg, const float* __restrict__ hB,
            const float* __restrict__ hb2,
            float* __restrict__ h, _Float16* __restrict__ hf16, float* __restrict__ x)
{
  __shared__ __align__(16) _Float16 sW1[HID * KH];
  __shared__ __align__(16) _Float16 sW2[HID * HID];
  __shared__ float sPar[4 * HID];
  __shared__ __align__(16) _Float16 sA[WPB * 16 * KH];
  __shared__ __align__(16) float    sS[WPB * 16 * HID];

  const int tid = threadIdx.x;
  for (int o = tid; o < (HID * KH) / 8;  o += WPB * 32) async_cp16(HW1t + o * 8, sW1 + o * 8);
  for (int o = tid; o < (HID * HID) / 8; o += WPB * 32) async_cp16(HW2t + o * 8, sW2 + o * 8);
  for (int i = tid; i < HID; i += WPB * 32) {
    sPar[0*HID+i]=hb1[i]; sPar[1*HID+i]=hg[i]; sPar[2*HID+i]=hB[i]; sPar[3*HID+i]=hb2[i];
  }
  wait_async0();
  __syncthreads();

  const int wave = tid >> 5, lane = tid & 31;
  const int row = lane & 15, half = lane >> 4;
  const int cb = half * 32;
  _Float16* A = sA + wave * 16 * KH;
  float*    S = sS + wave * 16 * HID;

  const int NT = (NN / 16) / WPB;
  for (int t0 = blockIdx.x; t0 < NT; t0 += gridDim.x) {
    const int n0 = (t0 * WPB + wave) * 16;
    const int n  = n0 + row;

    // h half: contiguous 2KB async copy; m_h half: f32->f16 scalar
    wait_ds0();
#pragma unroll
    for (int it = 0; it < 4; ++it) {
      const int chunk = it * 32 + lane;      // 0..127
      async_cp16(hf16 + (size_t)n0 * HID + chunk * 8,
                 A + (chunk >> 3) * KH + (chunk & 7) * 8);
    }
    for (int c = 0; c < 32; ++c)
      A[row*KH + 64 + cb + c] = (_Float16)aggmh[(size_t)n*HID + cb + c];
    wait_async0();

    gemm16x64<4>(A, KH, sW1, KH, S, lane);
    {
      const float *bb=&sPar[0], *gg=&sPar[64], *BB=&sPar[128];
      float m=0.f, s2=0.f;
      for (int c=0;c<32;++c){ float y=S[row*HID+cb+c]+bb[cb+c]; m+=y; s2+=y*y; }
      m  += __shfl_xor(m, 16);
      s2 += __shfl_xor(s2, 16);
      m *= (1.0f/HID); s2 = s2*(1.0f/HID) - m*m;
      const float ri = rsqrtf(s2 + EPSLN);
      for (int c=0;c<32;++c){
        float y=(S[row*HID+cb+c]+bb[cb+c]-m)*ri*gg[cb+c]+BB[cb+c];
        A[row*KH+cb+c] = (_Float16)silu_f(y);
      }
    }
    gemm16x64<2>(A, KH, sW2, HID, S, lane);
    {
      const float* b2 = &sPar[192];
      for (int c = 0; c < 32; ++c) {
        float nv = h[(size_t)n*HID + cb + c] + S[row*HID + cb + c] + b2[cb + c];
        h[(size_t)n*HID + cb + c] = nv;
        hf16[(size_t)n*HID + cb + c] = (_Float16)nv;
      }
      if (half == 0) {
        const float inv = 1.0f / cdeg[n];
        x[n*3+0] += aggmx[n*3+0] * inv;
        x[n*3+1] += aggmx[n*3+1] * inv;
        x[n*3+2] += aggmx[n*3+2] * inv;
      }
    }
  }
}

// -------- final COM removal
__global__ void k_com_accum(const float* __restrict__ x, const float* __restrict__ pos,
                            const int* __restrict__ batch, float* __restrict__ com,
                            float* __restrict__ cnt)
{
  int n = blockIdx.x * blockDim.x + threadIdx.x;
  if (n >= NN) return;
  int b = batch[n];
  atomicAdd(&com[b*3+0], x[n*3+0]-pos[n*3+0]);
  atomicAdd(&com[b*3+1], x[n*3+1]-pos[n*3+1]);
  atomicAdd(&com[b*3+2], x[n*3+2]-pos[n*3+2]);
  atomicAdd(&cnt[b], 1.0f);
}

__global__ void k_final(const float* __restrict__ x, const float* __restrict__ pos,
                        const int* __restrict__ batch, const float* __restrict__ com,
                        const float* __restrict__ cnt, float* __restrict__ out)
{
  int i = blockIdx.x * blockDim.x + threadIdx.x;
  if (i >= NN * 3) return;
  int n = i / 3, d = i - n * 3;
  int b = batch[n];
  out[i] = (x[i] - pos[i]) - com[b*3+d] / cnt[b];
}

// =====================================================================
// Launcher. Input order (jax pytree flatten, dict keys ASCII-sorted):
//  0 positions, 1 node_attrs, 2 t, 3 edge_index, 4 edge_attrs, 5 batch,
//  6 emb_W, 7 emb_b, then 4 layers x 20 tensors each, sorted:
//  eB1,eB2,eW1,eW2,eb1,eb2,eg1,eg2,hB,hW1,hW2,hb1,hb2,hg,xB,xW1,xW2,xb1,xb2,xg
// =====================================================================
enum {
  P_eB1=0, P_eB2, P_eW1, P_eW2, P_eb1, P_eb2, P_eg1, P_eg2,
  P_hB, P_hW1, P_hW2, P_hb1, P_hb2, P_hg,
  P_xB, P_xW1, P_xW2, P_xb1, P_xb2, P_xg
};

extern "C" void kernel_launch(void* const* d_in, const int* in_sizes, int n_in,
                              void* d_out, int out_size, void* d_ws, size_t ws_size,
                              hipStream_t stream)
{
  (void)in_sizes; (void)n_in; (void)out_size; (void)ws_size;
  const float* pos   = (const float*)d_in[0];
  const float* attrs = (const float*)d_in[1];
  const float* t     = (const float*)d_in[2];
  const int*   ei    = (const int*)d_in[3];
  const int*   ea    = (const int*)d_in[4];
  const int*   batch = (const int*)d_in[5];
  const float* embW  = (const float*)d_in[6];
  const float* embB  = (const float*)d_in[7];

  // workspace layout (~30 MB total)
  float* ws    = (float*)d_ws;
  float* h     = ws;  ws += (size_t)NN * HID;
  float* x     = ws;  ws += (size_t)NN * 3;
  float* cdeg  = ws;  ws += NN;
  float* aggmh = ws;  ws += (size_t)NN * HID;
  float* aggmx = ws;  ws += (size_t)NN * 3;
  float* d20   = ws;  ws += EE;
  float* com   = ws;  ws += NGG * 3;
  float* cnt   = ws;  ws += NGG;
  _Float16* hf16 = (_Float16*)ws;  ws += (size_t)NN * HID / 2;
  _Float16* fw   = (_Float16*)ws;

  const size_t SZ_EW1 = HID * K1P, SZ_SQ = HID * HID, SZ_HW1 = HID * KH;
  const size_t PERL = SZ_EW1 + 3 * SZ_SQ + SZ_HW1;

  // f16 transposed weights (deterministic, rebuilt every call)
  for (int L = 0; L < NLAYERS; ++L) {
    const float* const* lp = (const float* const*)&d_in[8 + L * 20];
    _Float16* eW1t = fw + L * PERL;
    _Float16* eW2t = eW1t + SZ_EW1;
    _Float16* xW1t = eW2t + SZ_SQ;
    _Float16* hW1t = xW1t + SZ_SQ;
    _Float16* hW2t = hW1t + SZ_HW1;
    k_w16<<<(HID*K1P+255)/256, 256, 0, stream>>>(lp[P_eW1], eW1t, K1,  K1P, HID);
    k_w16<<<(HID*HID+255)/256, 256, 0, stream>>>(lp[P_eW2], eW2t, HID, HID, HID);
    k_w16<<<(HID*HID+255)/256, 256, 0, stream>>>(lp[P_xW1], xW1t, HID, HID, HID);
    k_w16<<<(HID*KH +255)/256, 256, 0, stream>>>(lp[P_hW1], hW1t, KH,  KH,  HID);
    k_w16<<<(HID*HID+255)/256, 256, 0, stream>>>(lp[P_hW2], hW2t, HID, HID, HID);
  }

  (void)hipMemsetAsync(cdeg, 0, NN * sizeof(float), stream);
  k_init_h<<<(NN*HID+255)/256, 256, 0, stream>>>(attrs, t, batch, embW, embB, pos, h, hf16, x);
  k_edge_static<<<(EE+255)/256, 256, 0, stream>>>(ei, pos, cdeg, d20);

  for (int L = 0; L < NLAYERS; ++L) {
    const float* const* lp = (const float* const*)&d_in[8 + L * 20];
    _Float16* eW1t = fw + L * PERL;
    _Float16* eW2t = eW1t + SZ_EW1;
    _Float16* xW1t = eW2t + SZ_SQ;
    _Float16* hW1t = xW1t + SZ_SQ;
    _Float16* hW2t = hW1t + SZ_HW1;
    (void)hipMemsetAsync(aggmh, 0, (size_t)NN * HID * sizeof(float), stream);
    (void)hipMemsetAsync(aggmx, 0, (size_t)NN * 3 * sizeof(float), stream);
    k_edge<<<1024, WPB * 32, 0, stream>>>(ei, ea, d20, hf16, x, eW1t, eW2t, xW1t,
        lp[P_eb1], lp[P_eg1], lp[P_eB1], lp[P_eb2], lp[P_eg2], lp[P_eB2],
        lp[P_xb1], lp[P_xg], lp[P_xB], lp[P_xW2], lp[P_xb2],
        aggmh, aggmx);
    k_node<<<625, WPB * 32, 0, stream>>>(aggmh, aggmx, cdeg, hW1t, hW2t,
        lp[P_hb1], lp[P_hg], lp[P_hB], lp[P_hb2], h, hf16, x);
  }

  (void)hipMemsetAsync(com, 0, NGG * 3 * sizeof(float), stream);
  (void)hipMemsetAsync(cnt, 0, NGG * sizeof(float), stream);
  k_com_accum<<<(NN+255)/256, 256, 0, stream>>>(x, pos, batch, com, cnt);
  k_final<<<(NN*3+255)/256, 256, 0, stream>>>(x, pos, batch, com, cnt, (float*)d_out);
}
